// PytorchLlamaSDPA_43834436223586
// MI455X (gfx1250) — matrix operand order repaired and checked by reading, not compiled, not run on hardware
//
#include <hip/hip_runtime.h>

// ---------------- Problem constants ----------------
#define SEQ   2048
#define KV    2048
#define HD    128
#define NH    64
#define NKV   8
#define DIM   8192   // NH*HD

typedef __attribute__((ext_vector_type(16))) _Float16 v16h;
typedef __attribute__((ext_vector_type(8)))  float    v8f;

// ---------------- WMMA helpers ----------------
__device__ __forceinline__ v8f wmma32(v16h a, v16h b, v8f c) {
    // D = A(16x32 f16) x B(32x16 f16) + C(16x16 f32)
    return __builtin_amdgcn_wmma_f32_16x16x32_f16(
        /*neg_a=*/false, a, /*neg_b=*/false, b,
        /*c_mod=*/(short)0, c, /*reuse_a=*/false, /*reuse_b=*/false);
}

// A-fragment (16x32, 16-bit): lane = M row (lane%16). Per ISA layout:
// lanes 0-15: v0-3 -> K=0..7,   v4-7 -> K=16..23
// lanes16-31: v0-3 -> K=8..15,  v4-7 -> K=24..31
// => two b128 loads at rowptr + {8*half, 16+8*half}
__device__ __forceinline__ v16h load_a_frag(const _Float16* rowptr) {
    int hf = (threadIdx.x >> 4) & 1;
    union { int4 i[2]; v16h v; } u;
    u.i[0] = *(const int4*)(rowptr + hf * 8);
    u.i[1] = *(const int4*)(rowptr + 16 + hf * 8);
    return u.v;
}

// B-fragment (32x16, 16-bit): lane = N column (lane%16).
// same K order as A: lanes 0-15 K=0..7,16..23; lanes 16-31 K=8..15,24..31.
// => two b128 loads at rowptr + {8*half, 16+8*half}
__device__ __forceinline__ v16h load_b_frag(const _Float16* rowptr) {
    int hf = (threadIdx.x >> 4) & 1;
    union { int4 i[2]; v16h v; } u;
    u.i[0] = *(const int4*)(rowptr + hf * 8);
    u.i[1] = *(const int4*)(rowptr + 16 + hf * 8);
    return u.v;
}

// ---------------- Conversion kernels (fp32 -> f16 staging) ----------------
__global__ void cvt_q(const float* __restrict__ xq, _Float16* __restrict__ Q, int n) {
    int i = blockIdx.x * blockDim.x + threadIdx.x;
    if (i >= n) return;
    // xq: (s, h, 1, d) -> Q: [h][s][d]
    int d = i & (HD - 1);
    int h = (i >> 7) & (NH - 1);
    int s = i >> 13;
    Q[((size_t)h * SEQ + s) * HD + d] = (_Float16)xq[i];
}

__global__ void cvt_k(const float* __restrict__ keys, _Float16* __restrict__ K, int n) {
    int i = blockIdx.x * blockDim.x + threadIdx.x;
    if (i >= n) return;
    K[i] = (_Float16)keys[i];   // already [hkv][kv][d]
}

__global__ void cvt_v(const float* __restrict__ values, _Float16* __restrict__ Vt, int n) {
    int i = blockIdx.x * blockDim.x + threadIdx.x;
    if (i >= n) return;
    // values: [hkv][t][d] -> Vt: [hkv][d][t]
    int d = i & (HD - 1);
    int t = (i >> 7) & (KV - 1);
    int hk = i >> 18;
    Vt[((size_t)hk * HD + d) * KV + t] = (_Float16)values[i];
}

__global__ void cvt_w(const float* __restrict__ wo, _Float16* __restrict__ W, size_t n) {
    size_t i = (size_t)blockIdx.x * blockDim.x + threadIdx.x;
    if (i >= n) return;
    W[i] = (_Float16)wo[i];
}

// ---------------- Flash attention (one wave = 16 query rows) ----------------
__global__ __launch_bounds__(128) void attn_kernel(
    const _Float16* __restrict__ Qh, const _Float16* __restrict__ Kh,
    const _Float16* __restrict__ Vth, const float* __restrict__ mask,
    _Float16* __restrict__ attn)
{
    const int w    = threadIdx.x >> 5;          // wave id in WG (0..3)
    const int lane = threadIdx.x & 31;
    const int hf   = lane >> 4;                  // lane half (C-layout M offset)
    const int nl   = lane & 15;                  // N column / row-in-tile index
    const int head = blockIdx.x;
    const int q0   = blockIdx.y * 64 + w * 16;   // 16-row query tile base
    const int hkv  = head >> 3;                  // n_rep = 8

    // Q A-fragments for full head_dim (4 k-steps of 32), held in registers
    const _Float16* qrow = Qh + ((size_t)head * SEQ + q0 + nl) * HD;
    v16h qf[4];
#pragma unroll
    for (int st = 0; st < 4; ++st) qf[st] = load_a_frag(qrow + st * 32);

    const _Float16* Kbase = Kh  + (size_t)hkv * KV * HD;
    const _Float16* Vbase = Vth + (size_t)hkv * HD * KV;
    const float*    mrow  = mask + (size_t)(q0 + 8 * hf) * KV;  // rows r+8*hf

    float rm[8], rl[8];
    v8f   oacc[8] = {};
#pragma unroll
    for (int r = 0; r < 8; ++r) { rm[r] = -1e30f; rl[r] = 0.0f; }

    __shared__ __align__(16) _Float16 lds_p[4][16][32];  // wave-private P slab
    const float scale = 0.08838834764831845f;            // 1/sqrt(128)

    for (int kv0 = 0; kv0 < KV; kv0 += 32) {
        // ---- S = Q K^T for two 16-kv column tiles ----
        v8f s0 = {}, s1 = {};
        const _Float16* k0row = Kbase + (size_t)(kv0 + nl) * HD;
        const _Float16* k1row = k0row + (size_t)16 * HD;
#pragma unroll
        for (int st = 0; st < 4; ++st) {
            v16h kf0 = load_b_frag(k0row + st * 32);
            s0 = wmma32(qf[st], kf0, s0);
            v16h kf1 = load_b_frag(k1row + st * 32);
            s1 = wmma32(qf[st], kf1, s1);
        }

        // ---- scale + mask + online softmax (16-lane row reductions) ----
        float p0[8], p1[8], al[8];
#pragma unroll
        for (int r = 0; r < 8; ++r) {
            const float* mr = mrow + (size_t)r * KV + kv0;
            float a0 = s0[r] * scale + mr[nl];
            float a1 = s1[r] * scale + mr[16 + nl];
            float t = fmaxf(a0, a1);
            t = fmaxf(t, __shfl_xor(t, 1, 32));
            t = fmaxf(t, __shfl_xor(t, 2, 32));
            t = fmaxf(t, __shfl_xor(t, 4, 32));
            t = fmaxf(t, __shfl_xor(t, 8, 32));
            float nm = fmaxf(rm[r], t);
            al[r] = __expf(rm[r] - nm);
            rm[r] = nm;
            p0[r] = __expf(a0 - nm);
            p1[r] = __expf(a1 - nm);
            float rs = p0[r] + p1[r];
            rs += __shfl_xor(rs, 1, 32);
            rs += __shfl_xor(rs, 2, 32);
            rs += __shfl_xor(rs, 4, 32);
            rs += __shfl_xor(rs, 8, 32);
            rl[r] = rl[r] * al[r] + rs;
        }
#pragma unroll
        for (int c = 0; c < 8; ++c)
#pragma unroll
            for (int r = 0; r < 8; ++r) oacc[c][r] *= al[r];

        // ---- C-layout P -> A-fragment via wave-private LDS (in-order) ----
#pragma unroll
        for (int r = 0; r < 8; ++r) {
            lds_p[w][r + 8 * hf][nl]      = (_Float16)p0[r];
            lds_p[w][r + 8 * hf][16 + nl] = (_Float16)p1[r];
        }
        asm volatile("s_wait_dscnt 0" ::: "memory");
        v16h pf = load_a_frag(&lds_p[w][nl][0]);  // 16x32 A-fragment

        // ---- O += P * V  (V^T rows are contiguous over kv) ----
#pragma unroll
        for (int c = 0; c < 8; ++c) {
            v16h vf = load_b_frag(Vbase + (size_t)(c * 16 + nl) * KV + kv0);
            oacc[c] = wmma32(pf, vf, oacc[c]);
        }
    }

    // ---- normalize and write attn[s][head*128 + d] as f16 ----
    float inv[8];
#pragma unroll
    for (int r = 0; r < 8; ++r) inv[r] = 1.0f / rl[r];
#pragma unroll
    for (int c = 0; c < 8; ++c)
#pragma unroll
        for (int r = 0; r < 8; ++r) {
            size_t s = (size_t)(q0 + r + 8 * hf);
            attn[s * DIM + head * HD + c * 16 + nl] = (_Float16)(oacc[c][r] * inv[r]);
        }
}

// ---------------- out = attn (2048x8192 f16) x wo^T (f16, L2-resident) ----------------
// Block = 4 waves, all sharing the SAME 128-column W tile (WGP-cache broadcast),
// each wave owning a distinct 64-row M strip. Per wave: 64(M) x 128(N),
// 32 WMMAs per 32-k step, acc = 4x8 f32 fragments.
__global__ __launch_bounds__(128) void gemm_kernel(
    const _Float16* __restrict__ A, const _Float16* __restrict__ W,
    float* __restrict__ out)
{
    const int w    = threadIdx.x >> 5;
    const int lane = threadIdx.x & 31;
    const int hf   = lane >> 4;
    const int nl   = lane & 15;
    const int iT   = blockIdx.x & 63;            // 64 N-tiles of 128
    const int sS   = blockIdx.x >> 6;            // 8 M-supertiles of 256
    const int s0   = sS * 256 + w * 64;          // wave's 64-row M strip
    const int i0   = iT * 128;                   // block-shared N range

    v8f acc[4][8] = {};
    const _Float16* arow[4];
    const _Float16* wrow[8];
#pragma unroll
    for (int g = 0; g < 4; ++g) arow[g] = A + (size_t)(s0 + g * 16 + nl) * DIM;
#pragma unroll
    for (int c = 0; c < 8; ++c) wrow[c] = W + (size_t)(i0 + c * 16 + nl) * DIM;

    for (int j = 0; j < DIM; j += 32) {
        v16h af[4], bf[8];
#pragma unroll
        for (int g = 0; g < 4; ++g) af[g] = load_a_frag(arow[g] + j);
#pragma unroll
        for (int c = 0; c < 8; ++c) bf[c] = load_b_frag(wrow[c] + j);
        if (w == 0 && j + 64 < DIM) {            // W shared: one wave prefetches
#pragma unroll
            for (int c = 0; c < 8; ++c) __builtin_prefetch(wrow[c] + j + 64, 0, 1);
        }
#pragma unroll
        for (int g = 0; g < 4; ++g)
#pragma unroll
            for (int c = 0; c < 8; ++c) acc[g][c] = wmma32(af[g], bf[c], acc[g][c]);
    }

#pragma unroll
    for (int g = 0; g < 4; ++g)
#pragma unroll
        for (int c = 0; c < 8; ++c)
#pragma unroll
            for (int r = 0; r < 8; ++r) {
                size_t s = (size_t)(s0 + g * 16 + r + 8 * hf);
                out[s * DIM + i0 + c * 16 + nl] = acc[g][c][r];
            }
}

// ---------------- Host launch ----------------
extern "C" void kernel_launch(void* const* d_in, const int* in_sizes, int n_in,
                              void* d_out, int out_size, void* d_ws, size_t ws_size,
                              hipStream_t stream) {
    const float* xq     = (const float*)d_in[0];
    const float* keys   = (const float*)d_in[1];
    const float* values = (const float*)d_in[2];
    const float* mask   = (const float*)d_in[3];
    const float* wo     = (const float*)d_in[4];
    float* out = (float*)d_out;

    // f16 workspace layout
    const size_t QH_E = (size_t)NH * SEQ * HD;    // 16.8M
    const size_t KH_E = (size_t)NKV * KV * HD;    // 2.1M
    const size_t VT_E = KH_E;
    const size_t AT_E = (size_t)SEQ * DIM;        // 16.8M
    const size_t WO_E = (size_t)DIM * DIM;        // 67.1M
    _Float16* Qh  = (_Float16*)d_ws;
    _Float16* Kh  = Qh  + QH_E;
    _Float16* Vth = Kh  + KH_E;
    _Float16* At  = Vth + VT_E;
    _Float16* Wh  = At  + AT_E;

    // Stage 1: fp32 -> f16 conversions (+ layout transforms)
    cvt_q<<<(int)((QH_E + 255) / 256), 256, 0, stream>>>(xq, Qh, (int)QH_E);
    cvt_k<<<(int)((KH_E + 255) / 256), 256, 0, stream>>>(keys, Kh, (int)KH_E);
    cvt_v<<<(int)((VT_E + 255) / 256), 256, 0, stream>>>(values, Vth, (int)VT_E);
    cvt_w<<<(int)((WO_E + 255) / 256), 256, 0, stream>>>(wo, Wh, WO_E);

    // Stage 2: flash attention -> At (f16)
    attn_kernel<<<dim3(NH, SEQ / 64), 128, 0, stream>>>(Qh, Kh, Vth, mask, At);

    // Stage 3: output projection -> out (f32), 8 M-supertiles x 64 N-tiles
    gemm_kernel<<<512, 128, 0, stream>>>(At, Wh, out);
}
